// GQAttention_12678743458184
// MI455X (gfx1250) — matrix operand order repaired and checked by reading, not compiled, not run on hardware
//
#include <hip/hip_runtime.h>

// ---------------------------------------------------------------------------
// MI455X / gfx1250, wave32. All matmuls via V_WMMA_F32_16X16X32_BF16.
// ---------------------------------------------------------------------------

typedef __attribute__((ext_vector_type(16))) __bf16 v16bf;
typedef __attribute__((ext_vector_type(8)))  __bf16 v8bf;
typedef __attribute__((ext_vector_type(8)))  float  v8f;

// A-matrix fragment (16x32 bf16), contiguous-K order (not the ISA table order):
// lane m = lane&15 ; kb = (lane>=16)*8 ; elem i -> K=2*kb+i (matches frag_b16)
// => two contiguous 16-byte loads from this lane's row.
__device__ __forceinline__ v16bf frag_a16(const __bf16* rowp, int kb) {
  v8bf lo = *(const v8bf*)(rowp + 2 * kb);
  v8bf hi = *(const v8bf*)(rowp + 2 * kb + 8);
  v16bf f;
#pragma unroll
  for (int i = 0; i < 8; ++i) { f[i] = lo[i]; f[i + 8] = hi[i]; }
  return f;
}

// B-matrix fragment (32x16 bf16): lane n = lane&15 ; elem i -> K = kb16 + i,
// kb16 = (lane>=16)*16. colp must point at this lane's column stored K-contiguous,
// already offset by kb16 => two contiguous 16-byte loads.
__device__ __forceinline__ v16bf frag_b16(const __bf16* colp) {
  v8bf lo = *(const v8bf*)(colp);
  v8bf hi = *(const v8bf*)(colp + 8);
  v16bf f;
#pragma unroll
  for (int i = 0; i < 8; ++i) { f[i] = lo[i]; f[i + 8] = hi[i]; }
  return f;
}

// ---------------------------------------------------------------------------
// Kernel 1: fp32-in / fp32-out GEMM with bf16 WMMA core.
// C[M,N] = A[M,K] * B[K,N].  Block tile 128x128, 8 waves, each wave 32x64.
// K staged 32 at a time in LDS (converted to bf16), rows padded to 40 elems
// (80B) so every fragment load is 16B-aligned.
// ---------------------------------------------------------------------------
#define G_LDA 40

__global__ __launch_bounds__(256)
void gemm_bf16w_kernel(const float* __restrict__ A, const float* __restrict__ Bw,
                       float* __restrict__ C, int M, int N, int K) {
  __shared__ __align__(16) __bf16 As[128][G_LDA];   // [m][k]
  __shared__ __align__(16) __bf16 Bt[128][G_LDA];   // transposed: [n][k]

  const int tid  = threadIdx.x;
  const int lane = tid & 31;
  const int wave = tid >> 5;       // 0..7
  const int wm   = wave & 3;       // 4 sub-tiles of 32 rows
  const int wn   = wave >> 2;      // 2 sub-tiles of 64 cols
  const int m0   = blockIdx.y * 128;
  const int n0   = blockIdx.x * 128;

  v8f acc[2][4];
#pragma unroll
  for (int i = 0; i < 2; ++i)
#pragma unroll
    for (int j = 0; j < 4; ++j)
#pragma unroll
      for (int r = 0; r < 8; ++r) acc[i][j][r] = 0.0f;

  const int aRow = tid >> 1, aCol = (tid & 1) * 16;  // A tile: 128x32
  const int bK   = tid >> 3, bN   = (tid & 7) * 16;  // B tile: 32x128 (stored transposed)
  const int kb   = (lane >> 4) << 3;                 // A frag K base (0/8)
  const int kb16 = (lane >> 4) << 4;                 // B frag K base (0/16)
  const int ml   = lane & 15;

  for (int kt = 0; kt < K; kt += 32) {
    __syncthreads();
    {
      const float* ap = A + (size_t)(m0 + aRow) * K + kt + aCol;
#pragma unroll
      for (int i = 0; i < 16; ++i) As[aRow][aCol + i] = (__bf16)ap[i];
      const float* bp = Bw + (size_t)(kt + bK) * N + n0 + bN;
#pragma unroll
      for (int i = 0; i < 16; ++i) Bt[bN + i][bK] = (__bf16)bp[i];
    }
    __syncthreads();

    v16bf af[2], bf[4];
#pragma unroll
    for (int mi = 0; mi < 2; ++mi)
      af[mi] = frag_a16(&As[wm * 32 + mi * 16 + ml][0], kb);
#pragma unroll
    for (int ni = 0; ni < 4; ++ni)
      bf[ni] = frag_b16(&Bt[wn * 64 + ni * 16 + ml][kb16]);
#pragma unroll
    for (int mi = 0; mi < 2; ++mi)
#pragma unroll
      for (int ni = 0; ni < 4; ++ni)
        acc[mi][ni] = __builtin_amdgcn_wmma_f32_16x16x32_bf16(
            false, af[mi], false, bf[ni], (short)0, acc[mi][ni], false, false);
  }

  // Epilogue: C/D layout — lane n = lane&15, elem r -> row r + 8*(lane>=16).
  const int rbase = (lane >> 4) << 3;
#pragma unroll
  for (int mi = 0; mi < 2; ++mi)
#pragma unroll
    for (int ni = 0; ni < 4; ++ni)
#pragma unroll
      for (int r = 0; r < 8; ++r) {
        int row = m0 + wm * 32 + mi * 16 + rbase + r;
        int col = n0 + wn * 64 + ni * 16 + ml;
        C[(size_t)row * N + col] = acc[mi][ni][r];
      }
}

// ---------------------------------------------------------------------------
// Kernel 2: RoPE (interleaved pairs) + repack to head-major bf16, plus
// optional fp32 cache output (k_cache / v_cache).
// src: (B*S, nh*128) fp32 ; dst: (B, nh, S, 128) bf16 ; cache same shape fp32.
// ---------------------------------------------------------------------------
__global__ __launch_bounds__(256)
void rope_pack_kernel(const float* __restrict__ src, const float* __restrict__ fcos,
                      const float* __restrict__ fsin, __bf16* __restrict__ dst,
                      float* __restrict__ cache, int nh, int S, int do_rope) {
  int idx = blockIdx.x * blockDim.x + threadIdx.x;
  int hd2 = idx & 63;
  int h   = (idx >> 6) % nh;
  int t   = idx / (64 * nh);       // = b*S + s
  int s   = t % S;
  int b   = t / S;

  const float* p = src + (size_t)t * (nh * 128) + h * 128 + 2 * hd2;
  float xr = p[0], xi = p[1];
  if (do_rope) {
    float c = fcos[s * 64 + hd2], sn = fsin[s * 64 + hd2];
    float orr = xr * c - xi * sn;
    float oi  = xr * sn + xi * c;
    xr = orr; xi = oi;
  }
  size_t di = (((size_t)b * nh + h) * S + s) * 128 + 2 * hd2;
  dst[di]     = (__bf16)xr;
  dst[di + 1] = (__bf16)xi;
  if (cache) { cache[di] = xr; cache[di + 1] = xi; }
}

// ---------------------------------------------------------------------------
// Kernel 3: causal flash attention with GQA. Block = 128 thr (4 waves),
// one block per (b, head, 64 q-rows); each wave owns 16 q-rows.
// Per 32-key step: S = Q*K^T (8 WMMA), online softmax (shfl_xor row
// reductions within 16-lane halves), P transposed through wave-private LDS,
// O += P*V (8 WMMA).
// ---------------------------------------------------------------------------
#define A_LDK 136   // 128 + 8 pad (272B rows, 16B aligned)
#define A_LDV 40    // 32 + 8 pad (80B rows)

__global__ __launch_bounds__(128)
void attn_kernel(const __bf16* __restrict__ qbf, const __bf16* __restrict__ kbf,
                 const __bf16* __restrict__ vbf, float* __restrict__ attn_out,
                 int B, int S, int NH, int NKV) {
  __shared__ __align__(16) __bf16 Ks[32][A_LDK];    // K tile  [key][hd]
  __shared__ __align__(16) __bf16 Vt[128][A_LDV];   // V tile  [hd][key] (transposed)
  __shared__ __align__(16) __bf16 Ps[4][16][A_LDV]; // wave-private P [row][key]

  const int tid  = threadIdx.x;
  const int lane = tid & 31;
  const int wave = tid >> 5;
  const int q0   = blockIdx.x * 64;
  const int h    = blockIdx.y;
  const int b    = blockIdx.z;
  const int kvh  = h / (NH / NKV);
  const int qr   = q0 + wave * 16;
  const int ml   = lane & 15;
  const int kb8  = (lane >> 4) << 3;
  const int kb16 = (lane >> 4) << 4;

  // Q fragments for this wave's 16 rows (loaded once; 4 k-chunks of 32 over HD=128)
  const __bf16* qbase = qbf + (((size_t)b * NH + h) * S + qr) * 128;
  v16bf qf[4];
#pragma unroll
  for (int j = 0; j < 4; ++j)
    qf[j] = frag_a16(qbase + (size_t)ml * 128 + j * 32, kb8);

  v8f o[8];
  float mstate[8], lstate[8];
#pragma unroll
  for (int t = 0; t < 8; ++t)
#pragma unroll
    for (int r = 0; r < 8; ++r) o[t][r] = 0.0f;
#pragma unroll
  for (int r = 0; r < 8; ++r) { mstate[r] = -3.0e38f; lstate[r] = 0.0f; }

  const size_t kvbase  = ((size_t)b * NKV + kvh) * (size_t)S * 128;
  const float  scale   = 0.08838834764831845f;  // 1/sqrt(128)
  const int    nblocks = q0 / 32 + 2;           // causal bound (uniform per block)

  for (int kvb = 0; kvb < nblocks; ++kvb) {
    const int kv0 = kvb * 32;
    __syncthreads();  // prior iteration's LDS reads complete
    {
      // K tile: 32 keys x 128 hd, row-major (hd contiguous for QK^T B-frags)
      int row = tid >> 2, cb = (tid & 3) * 32;
      const __bf16* src = kbf + kvbase + (size_t)(kv0 + row) * 128 + cb;
#pragma unroll
      for (int i = 0; i < 4; ++i)
        *(v8bf*)&Ks[row][cb + 8 * i] = *(const v8bf*)(src + 8 * i);
      // V tile transposed: Vt[hd][key] (key contiguous for PV B-frags)
      int key = tid & 31, hb = (tid >> 5) * 32;
      const __bf16* vsrc = vbf + kvbase + (size_t)(kv0 + key) * 128 + hb;
#pragma unroll
      for (int i = 0; i < 32; ++i) Vt[hb + i][key] = vsrc[i];
    }
    __syncthreads();

    // S = Q * K^T : 16 q-rows x 32 keys (two 16-key n-tiles), contraction HD=128
    v8f s0, s1;
#pragma unroll
    for (int r = 0; r < 8; ++r) { s0[r] = 0.0f; s1[r] = 0.0f; }
#pragma unroll
    for (int j = 0; j < 4; ++j) {
      v16bf b0 = frag_b16(&Ks[ml][j * 32 + kb16]);
      v16bf b1 = frag_b16(&Ks[16 + ml][j * 32 + kb16]);
      s0 = __builtin_amdgcn_wmma_f32_16x16x32_bf16(false, qf[j], false, b0, (short)0, s0, false, false);
      s1 = __builtin_amdgcn_wmma_f32_16x16x32_bf16(false, qf[j], false, b1, (short)0, s1, false, false);
    }

    // Online softmax (causal mask applied analytically).
    const int keyc = kv0 + ml;
#pragma unroll
    for (int r = 0; r < 8; ++r) {
      int   qrow = qr + kb8 + r;
      float v0 = (keyc      <= qrow) ? s0[r] * scale : -1.0e30f;
      float v1 = (keyc + 16 <= qrow) ? s1[r] * scale : -1.0e30f;
      float mx = fmaxf(v0, v1);
#pragma unroll
      for (int off = 1; off < 16; off <<= 1) mx = fmaxf(mx, __shfl_xor(mx, off, 32));
      float mnew  = fmaxf(mstate[r], mx);
      float alpha = __expf(mstate[r] - mnew);
      float p0 = __expf(v0 - mnew), p1 = __expf(v1 - mnew);
      float rs = p0 + p1;
#pragma unroll
      for (int off = 1; off < 16; off <<= 1) rs += __shfl_xor(rs, off, 32);
      lstate[r] = lstate[r] * alpha + rs;
      mstate[r] = mnew;
#pragma unroll
      for (int t = 0; t < 8; ++t) o[t][r] *= alpha;
      Ps[wave][kb8 + r][ml]      = (__bf16)p0;   // C-layout -> row-major P in LDS
      Ps[wave][kb8 + r][16 + ml] = (__bf16)p1;
    }
    __syncthreads();  // Ps visible (also keeps barrier count uniform)

    // O += P * V : contraction over 32 keys, 8 hd n-tiles
    v16bf pf = frag_a16(&Ps[wave][ml][0], kb8);
#pragma unroll
    for (int t = 0; t < 8; ++t) {
      v16bf vf = frag_b16(&Vt[t * 16 + ml][kb16]);
      o[t] = __builtin_amdgcn_wmma_f32_16x16x32_bf16(false, pf, false, vf, (short)0, o[t], false, false);
    }
  }

  // Normalize and write attn output in (B*S, NH*128) row-major fp32.
#pragma unroll
  for (int r = 0; r < 8; ++r) {
    float  inv = 1.0f / lstate[r];
    size_t row = (size_t)b * S + qr + kb8 + r;
#pragma unroll
    for (int t = 0; t < 8; ++t)
      attn_out[row * (size_t)(NH * 128) + h * 128 + t * 16 + ml] = o[t][r] * inv;
  }
}

// ---------------------------------------------------------------------------
// Host launcher
// ---------------------------------------------------------------------------
extern "C" void kernel_launch(void* const* d_in, const int* in_sizes, int n_in,
                              void* d_out, int out_size, void* d_ws, size_t ws_size,
                              hipStream_t stream) {
  (void)in_sizes; (void)n_in; (void)out_size; (void)ws_size;
  const float* x  = (const float*)d_in[0];
  const float* wq = (const float*)d_in[1];
  const float* wk = (const float*)d_in[2];
  const float* wv = (const float*)d_in[3];
  const float* wo = (const float*)d_in[4];
  const float* fc = (const float*)d_in[5];
  const float* fs = (const float*)d_in[6];
  // d_in[7] (mask) unused: causal mask applied analytically in attn_kernel.

  const int B = 2, S = 2048, H = 2048, NH = 16, NKV = 4, HD = 128;
  const int M = B * S;                 // 4096
  const int NKVD = NKV * HD;           // 512

  char* w = (char*)d_ws;
  float*  qf = (float*)w;   w += (size_t)M * H    * sizeof(float);
  float*  kf = (float*)w;   w += (size_t)M * NKVD * sizeof(float);
  float*  vf = (float*)w;   w += (size_t)M * NKVD * sizeof(float);
  __bf16* qb = (__bf16*)w;  w += (size_t)M * H    * sizeof(__bf16);
  __bf16* kb = (__bf16*)w;  w += (size_t)M * NKVD * sizeof(__bf16);
  __bf16* vb = (__bf16*)w;  w += (size_t)M * NKVD * sizeof(__bf16);
  float*  attn = qf;  // qf consumed by rope before attention writes here

  float* out_p   = (float*)d_out;
  float* k_cache = out_p + (size_t)B * S * H;
  float* v_cache = k_cache + (size_t)B * NKV * S * HD;

  dim3 blk256(256);
  // QKV projections (bf16 WMMA GEMM)
  gemm_bf16w_kernel<<<dim3(H / 128,    M / 128), blk256, 0, stream>>>(x, wq, qf, M, H,    H);
  gemm_bf16w_kernel<<<dim3(NKVD / 128, M / 128), blk256, 0, stream>>>(x, wk, kf, M, NKVD, H);
  gemm_bf16w_kernel<<<dim3(NKVD / 128, M / 128), blk256, 0, stream>>>(x, wv, vf, M, NKVD, H);

  // RoPE + repack (and emit fp32 k/v caches)
  rope_pack_kernel<<<(unsigned)((size_t)M * NH  * 64 / 256), blk256, 0, stream>>>(qf, fc, fs, qb, nullptr,  NH,  S, 1);
  rope_pack_kernel<<<(unsigned)((size_t)M * NKV * 64 / 256), blk256, 0, stream>>>(kf, fc, fs, kb, k_cache, NKV, S, 1);
  rope_pack_kernel<<<(unsigned)((size_t)M * NKV * 64 / 256), blk256, 0, stream>>>(vf, fc, fs, vb, v_cache, NKV, S, 0);

  // Causal GQA flash attention
  attn_kernel<<<dim3(S / 64, NH, B), dim3(128), 0, stream>>>(qb, kb, vb, attn, B, S, NH, NKV);

  // Output projection
  gemm_bf16w_kernel<<<dim3(H / 128, M / 128), blk256, 0, stream>>>(attn, wo, out_p, M, H, H);
}
